// QuadMambaBlock_13280038879659
// MI455X (gfx1250) — compile-verified, matrix operand order checked by reading, not run on hardware
//
#include <hip/hip_runtime.h>
#include <hip/hip_bf16.h>

// ---------------------------------------------------------------------------
// QuadMamba block for MI455X (gfx1250, wave32, WMMA).
//   GEMMs (in_proj / out_proj / fuse) -> v_wmma_f32_16x16x32_bf16,
//     each wave computes 1 M-tile x 4 N-tiles; all 4 B fragments of a K-step
//     are held live so VMEM latency overlaps the WMMA quartet.
//   conv / x_proj / dt_proj -> VALU kernels.
//   selective scan -> 4-step chunks, double-buffered LDS stage for B/C,
//     one barrier per chunk, batched global loads for latency overlap.
// ---------------------------------------------------------------------------

typedef __attribute__((ext_vector_type(16))) __bf16 v16bf;
typedef __attribute__((ext_vector_type(8)))  __bf16 v8bf;
typedef __attribute__((ext_vector_type(8)))  float  v8f;

static constexpr int BATCH = 2;
static constexpr int LSEQ  = 4096;   // 64*64
static constexpr int HW    = 64;
static constexpr int DM    = 192;    // d_model
static constexpr int DIN   = 384;    // d_inner
static constexpr int NXZ   = 768;    // 2*d_inner
static constexpr int DST   = 16;     // d_state
static constexpr int DTR   = 12;     // dt_rank
static constexpr int NTOK  = BATCH * LSEQ;       // 8192
static constexpr int MT    = NTOK / 16;          // 512 M-tiles

union BFrag { v16bf v; v8bf h[2]; };

__device__ __forceinline__ float dev_silu(float x) {
    return x / (1.0f + __expf(-x));
}

// -------------------------------- prep -------------------------------------

__global__ void k_cvt_bf16(const float* __restrict__ src, __bf16* __restrict__ dst, int n) {
    int i = blockIdx.x * 256 + threadIdx.x;
    if (i < n) dst[i] = (__bf16)src[i];
}

// LayerNorm: one wave per token row, 6 elems/lane, shuffle reduction.
__global__ void k_layernorm(const float* __restrict__ x,
                            const float* __restrict__ g,
                            const float* __restrict__ b,
                            __bf16* __restrict__ xn) {
    int row  = blockIdx.x;
    int lane = threadIdx.x;           // 0..31
    const float* xr = x + (size_t)row * DM;
    float v[6], s = 0.f, s2 = 0.f;
    #pragma unroll
    for (int i = 0; i < 6; ++i) {
        v[i] = xr[lane + 32 * i];
        s  += v[i];
        s2 += v[i] * v[i];
    }
    #pragma unroll
    for (int o = 16; o > 0; o >>= 1) {
        s  += __shfl_xor(s,  o, 32);
        s2 += __shfl_xor(s2, o, 32);
    }
    float mu  = s * (1.0f / DM);
    float var = s2 * (1.0f / DM) - mu * mu;
    float inv = rsqrtf(var + 1e-5f);
    #pragma unroll
    for (int i = 0; i < 6; ++i) {
        int c = lane + 32 * i;
        xn[(size_t)row * DM + c] = (__bf16)((v[i] - mu) * inv * g[c] + b[c]);
    }
}

// ------------------------------ WMMA GEMMs ----------------------------------
// 96 threads = 3 waves; each wave computes one 16-row M-tile x four 16-col
// N-tiles. Per K-step: 1 A fragment + 4 live B fragments -> 4 WMMAs whose
// VMEM waits are staggered (loadcnt 6/4/2/0) instead of serialized.

__device__ __forceinline__ void gemm_kstep(const __bf16* __restrict__ arow,
                                           const __bf16* const* brow,
                                           int k, int koff, v8f* acc) {
    BFrag a;
    a.h[0] = *(const v8bf*)(arow + k + koff);
    a.h[1] = *(const v8bf*)(arow + k + 16 + koff);
    BFrag bm[4];
    #pragma unroll
    for (int j = 0; j < 4; ++j) {
        bm[j].h[0] = *(const v8bf*)(brow[j] + k + koff);
        bm[j].h[1] = *(const v8bf*)(brow[j] + k + 16 + koff);
    }
    #pragma unroll
    for (int j = 0; j < 4; ++j)
        acc[j] = __builtin_amdgcn_wmma_f32_16x16x32_bf16(false, a.v, false, bm[j].v,
                                                         (short)0, acc[j], false, false);
}

// in_proj with input-row permutation per direction: xz[tok][0..767]
__global__ __launch_bounds__(96)
void k_wmma_inproj(const __bf16* __restrict__ xn, const __bf16* __restrict__ W,
                   float* __restrict__ xz, int dir) {
    int lane   = threadIdx.x & 31;
    int wv     = threadIdx.x >> 5;                 // 0..2
    int mtile  = blockIdx.x;
    int ntbase = (blockIdx.y * 3 + wv) * 4;        // 4 N-tiles per wave
    int mrow   = mtile * 16 + (lane & 15);         // flat token index
    int bidx   = mrow >> 12;                       // / LSEQ
    int l      = mrow & (LSEQ - 1);
    int src;
    if (dir == 0)      src = l;
    else if (dir == 1) src = LSEQ - 1 - l;
    else if (dir == 2) { int r = l >> 6, c = l & (HW - 1); src = (r << 6) + (HW - 1 - c); }
    else               { int r = l >> 6, c = l & (HW - 1); src = ((HW - 1 - r) << 6) + c; }
    const __bf16* arow = xn + (size_t)(bidx * LSEQ + src) * DM;
    const __bf16* brow[4];
    #pragma unroll
    for (int j = 0; j < 4; ++j)
        brow[j] = W + (size_t)((ntbase + j) * 16 + (lane & 15)) * DM;
    int koff = (lane >> 4) * 8;
    v8f acc[4] = {};
    #pragma unroll
    for (int k = 0; k < DM; k += 32)
        gemm_kstep(arow, brow, k, koff, acc);
    int mbase = mtile * 16 + (lane >> 4) * 8;
    #pragma unroll
    for (int j = 0; j < 4; ++j) {
        int n = (ntbase + j) * 16 + (lane & 15);
        #pragma unroll
        for (int r = 0; r < 8; ++r)
            xz[(size_t)(mbase + r) * NXZ + n] = acc[j][r];
    }
}

// out_proj: y_act[8192x384]bf16 x W[192x384] -> bf16 into concat cols [col0, col0+192)
__global__ __launch_bounds__(96)
void k_wmma_outproj(const __bf16* __restrict__ yact, const __bf16* __restrict__ W,
                    __bf16* __restrict__ concat, int col0) {
    int lane   = threadIdx.x & 31;
    int wv     = threadIdx.x >> 5;
    int mtile  = blockIdx.x;
    int ntbase = wv * 4;                           // 12 N-tiles total
    const __bf16* arow = yact + (size_t)(mtile * 16 + (lane & 15)) * DIN;
    const __bf16* brow[4];
    #pragma unroll
    for (int j = 0; j < 4; ++j)
        brow[j] = W + (size_t)((ntbase + j) * 16 + (lane & 15)) * DIN;
    int koff = (lane >> 4) * 8;
    v8f acc[4] = {};
    #pragma unroll
    for (int k = 0; k < DIN; k += 32)
        gemm_kstep(arow, brow, k, koff, acc);
    int mbase = mtile * 16 + (lane >> 4) * 8;
    #pragma unroll
    for (int j = 0; j < 4; ++j) {
        int n = col0 + (ntbase + j) * 16 + (lane & 15);
        #pragma unroll
        for (int r = 0; r < 8; ++r)
            concat[(size_t)(mbase + r) * NXZ + n] = (__bf16)acc[j][r];
    }
}

// fuse: concat[8192x768]bf16 x fuse_w[192x768] + bias -> out = x + silu(.)
__global__ __launch_bounds__(96)
void k_wmma_fuse(const __bf16* __restrict__ concat, const __bf16* __restrict__ W,
                 const float* __restrict__ fb, const float* __restrict__ x,
                 float* __restrict__ out) {
    int lane   = threadIdx.x & 31;
    int wv     = threadIdx.x >> 5;
    int mtile  = blockIdx.x;
    int ntbase = wv * 4;                           // 12 N-tiles total
    const __bf16* arow = concat + (size_t)(mtile * 16 + (lane & 15)) * NXZ;
    const __bf16* brow[4];
    #pragma unroll
    for (int j = 0; j < 4; ++j)
        brow[j] = W + (size_t)((ntbase + j) * 16 + (lane & 15)) * NXZ;
    int koff = (lane >> 4) * 8;
    v8f acc[4] = {};
    #pragma unroll
    for (int k = 0; k < NXZ; k += 32)
        gemm_kstep(arow, brow, k, koff, acc);
    int mbase = mtile * 16 + (lane >> 4) * 8;
    #pragma unroll
    for (int j = 0; j < 4; ++j) {
        int n = (ntbase + j) * 16 + (lane & 15);
        float bias = fb[n];
        #pragma unroll
        for (int r = 0; r < 8; ++r) {
            size_t idx = (size_t)(mbase + r) * DM + n;
            float v = acc[j][r] + bias;
            out[idx] = x[idx] + dev_silu(v);
        }
    }
}

// ------------------------- conv / x_proj / dt --------------------------------

__global__ void k_conv_silu(const float* __restrict__ xz, const float* __restrict__ cw,
                            const float* __restrict__ cb, float* __restrict__ u_act) {
    int idx = blockIdx.x * 256 + threadIdx.x;     // over NTOK*DIN
    if (idx >= NTOK * DIN) return;
    int e = idx % DIN;
    int t = (idx / DIN) & (LSEQ - 1);
    int b = idx / (DIN * LSEQ);
    float acc = cb[e];
    #pragma unroll
    for (int j = 0; j < 4; ++j) {
        int tt = t - 3 + j;
        if (tt >= 0)
            acc += cw[e * 4 + j] * xz[(size_t)(b * LSEQ + tt) * NXZ + e];
    }
    u_act[idx] = dev_silu(acc);
}

// one token per block, 384 threads.
// Stage 1: 44 dot-products of length 384, each split across 8 lanes
//          (xor-shuffle reduction stays inside a wave32).
// Stage 2: dt = softplus(dt_proj) for all 384 channels.
__global__ __launch_bounds__(384)
void k_xproj_dt(const float* __restrict__ u_act, const float* __restrict__ xpw,
                const float* __restrict__ dtw, const float* __restrict__ dtb,
                float* __restrict__ dt, float* __restrict__ Bc, float* __restrict__ Cc) {
    __shared__ float urow[DIN];
    __shared__ float xd[DTR + 2 * DST];           // 44
    int tok = blockIdx.x;
    int tid = threadIdx.x;
    urow[tid] = u_act[(size_t)tok * DIN + tid];
    __syncthreads();
    if (tid < (DTR + 2 * DST) * 8) {              // 352 threads
        int f    = tid >> 3;
        int part = tid & 7;
        const float* wr = xpw + f * DIN + part * 48;
        const float* ur = urow + part * 48;
        float a = 0.f;
        #pragma unroll 8
        for (int k = 0; k < 48; ++k) a += wr[k] * ur[k];
        #pragma unroll
        for (int o = 1; o < 8; o <<= 1) a += __shfl_xor(a, o, 32);
        if (part == 0) xd[f] = a;
    }
    __syncthreads();
    float a = dtb[tid];
    #pragma unroll
    for (int r = 0; r < DTR; ++r) a += dtw[tid * DTR + r] * xd[r];
    float sp = (a > 20.f) ? a : logf(1.f + __expf(a));
    dt[(size_t)tok * DIN + tid] = sp;
    if (tid < DST)           Bc[(size_t)tok * DST + tid]         = xd[DTR + tid];
    else if (tid < 2 * DST)  Cc[(size_t)tok * DST + (tid - DST)] = xd[DTR + DST + (tid - DST)];
}

// ------------------------------ selective scan -------------------------------
// grid (BATCH, 6): 64 channels per block; 16-float state per channel in regs.
// Processes 4 timesteps per chunk: one barrier per chunk, double-buffered LDS
// for B/C, all global loads for the chunk issued up front (MLP >= 12).
static constexpr int SCHUNK = 4;

__global__ __launch_bounds__(64)
void k_scan(const float* __restrict__ dt, const float* __restrict__ u_act,
            const float* __restrict__ Bc, const float* __restrict__ Cc,
            const float* __restrict__ xz, const float* __restrict__ A_log,
            const float* __restrict__ Dp, __bf16* __restrict__ y_act) {
    int b   = blockIdx.x;
    int e   = blockIdx.y * 64 + threadIdx.x;      // channel
    int tid = threadIdx.x;
    float A[DST], h[DST];
    #pragma unroll
    for (int s = 0; s < DST; ++s) {
        A[s] = -__expf(A_log[(size_t)e * DST + s]);
        h[s] = 0.f;
    }
    float Dv = Dp[e];
    __shared__ float sBC[2][SCHUNK][2 * DST];     // [buf][step][B(16)|C(16)]
    for (int t0 = 0; t0 < LSEQ; t0 += SCHUNK) {
        int buf = (t0 >> 2) & 1;
        size_t tok0 = (size_t)b * LSEQ + t0;
        // per-thread chunk loads (independent of LDS; overlap with staging)
        float d4[SCHUNK], u4[SCHUNK], z4[SCHUNK];
        #pragma unroll
        for (int s = 0; s < SCHUNK; ++s) {
            d4[s] = dt[(tok0 + s) * DIN + e];
            u4[s] = u_act[(tok0 + s) * DIN + e];
            z4[s] = xz[(tok0 + s) * NXZ + DIN + e];
        }
        // stage B/C for 4 steps: 128 floats, 2 per thread
        #pragma unroll
        for (int i = 0; i < 2; ++i) {
            int flat = tid + i * 64;
            int st   = flat >> 5;
            int j    = flat & 31;
            sBC[buf][st][j] = (j < DST) ? Bc[(tok0 + st) * DST + j]
                                        : Cc[(tok0 + st) * DST + (j - DST)];
        }
        __syncthreads();
        #pragma unroll
        for (int s4 = 0; s4 < SCHUNK; ++s4) {
            float d  = d4[s4];
            float du = d * u4[s4];
            float y  = 0.f;
            #pragma unroll
            for (int s = 0; s < DST; ++s) {
                h[s] = h[s] * __expf(d * A[s]) + du * sBC[buf][s4][s];
                y   += h[s] * sBC[buf][s4][DST + s];
            }
            y += u4[s4] * Dv;
            y *= dev_silu(z4[s4]);
            y_act[(tok0 + s4) * DIN + e] = (__bf16)y;
        }
    }
}

// --------------------------------- launch ------------------------------------

extern "C" void kernel_launch(void* const* d_in, const int* in_sizes, int n_in,
                              void* d_out, int out_size, void* d_ws, size_t ws_size,
                              hipStream_t stream) {
    const float* x        = (const float*)d_in[0];
    const float* in_w     = (const float*)d_in[3];   // [4,768,192]
    const float* conv_w   = (const float*)d_in[4];   // [4,384,4]
    const float* conv_b   = (const float*)d_in[5];   // [4,384]
    const float* xproj_w  = (const float*)d_in[6];   // [4,44,384]
    const float* dtproj_w = (const float*)d_in[7];   // [4,384,12]
    const float* dtproj_b = (const float*)d_in[8];   // [4,384]
    const float* A_log    = (const float*)d_in[9];   // [4,384,16]
    const float* D_param  = (const float*)d_in[10];  // [4,384]
    const float* out_w    = (const float*)d_in[11];  // [4,192,384]
    const float* ln_g     = (const float*)d_in[12];
    const float* ln_b     = (const float*)d_in[13];
    const float* fuse_w   = (const float*)d_in[14];  // [192,768]
    const float* fuse_b   = (const float*)d_in[15];
    float* out = (float*)d_out;

    // workspace carve-up (256B aligned); per-direction buffers are reused
    char* ws = (char*)d_ws;
    size_t off = 0;
    auto carve = [&](size_t bytes) {
        char* p = ws + off;
        off = (off + bytes + 255) & ~(size_t)255;
        return p;
    };
    float*  xz      = (float*) carve((size_t)NTOK * NXZ * 4);   // 25.2 MB
    float*  u_act   = (float*) carve((size_t)NTOK * DIN * 4);   // 12.6 MB
    float*  dt_buf  = (float*) carve((size_t)NTOK * DIN * 4);   // 12.6 MB
    float*  Bc      = (float*) carve((size_t)NTOK * DST * 4);
    float*  Cc      = (float*) carve((size_t)NTOK * DST * 4);
    __bf16* y_act   = (__bf16*)carve((size_t)NTOK * DIN * 2);   // 6.3 MB
    __bf16* concat  = (__bf16*)carve((size_t)NTOK * NXZ * 2);   // 12.6 MB
    __bf16* xn_bf   = (__bf16*)carve((size_t)NTOK * DM * 2);    // 3.1 MB
    __bf16* win_bf  = (__bf16*)carve((size_t)4 * NXZ * DM * 2);
    __bf16* wout_bf = (__bf16*)carve((size_t)4 * DM * DIN * 2);
    __bf16* wfus_bf = (__bf16*)carve((size_t)DM * NXZ * 2);
    (void)ws_size;

    // weight conversion to bf16
    {
        int n1 = 4 * NXZ * DM, n2 = 4 * DM * DIN, n3 = DM * NXZ;
        k_cvt_bf16<<<(n1 + 255) / 256, 256, 0, stream>>>(in_w,   win_bf,  n1);
        k_cvt_bf16<<<(n2 + 255) / 256, 256, 0, stream>>>(out_w,  wout_bf, n2);
        k_cvt_bf16<<<(n3 + 255) / 256, 256, 0, stream>>>(fuse_w, wfus_bf, n3);
    }

    // LayerNorm -> bf16 xn
    k_layernorm<<<NTOK, 32, 0, stream>>>(x, ln_g, ln_b, xn_bf);

    for (int dir = 0; dir < 4; ++dir) {
        // in_proj (WMMA, permuted gather): 48 N-tiles = grid.y(4) * 3 waves * 4
        k_wmma_inproj<<<dim3(MT, 4), 96, 0, stream>>>(
            xn_bf, win_bf + (size_t)dir * NXZ * DM, xz, dir);

        // depthwise causal conv + silu
        k_conv_silu<<<(NTOK * DIN + 255) / 256, 256, 0, stream>>>(
            xz, conv_w + dir * DIN * 4, conv_b + dir * DIN, u_act);

        // x_proj (44 outs) + dt_proj + softplus
        k_xproj_dt<<<NTOK, DIN, 0, stream>>>(
            u_act, xproj_w + dir * (DTR + 2 * DST) * DIN,
            dtproj_w + dir * DIN * DTR, dtproj_b + dir * DIN,
            dt_buf, Bc, Cc);

        // selective scan + gate + D skip
        k_scan<<<dim3(BATCH, DIN / 64), 64, 0, stream>>>(
            dt_buf, u_act, Bc, Cc, xz,
            A_log + (size_t)dir * DIN * DST, D_param + dir * DIN, y_act);

        // out_proj (WMMA) -> concat columns [dir*192, dir*192+192)
        k_wmma_outproj<<<dim3(MT, 1), 96, 0, stream>>>(
            y_act, wout_bf + (size_t)dir * DM * DIN, concat, dir * DM);
    }

    // fuse GEMM (WMMA) + bias + silu + residual -> d_out
    k_wmma_fuse<<<dim3(MT, 1), 96, 0, stream>>>(concat, wfus_bf, fuse_b, x, out);
}
